// ModulatedConv2d_72507637891586
// MI455X (gfx1250) — matrix operand order repaired and checked by reading, not compiled, not run on hardware
//
#include <hip/hip_runtime.h>

typedef __attribute__((ext_vector_type(16))) _Float16 v16h;
typedef __attribute__((ext_vector_type(8)))  float    v8f;

// ---------------------------------------------------------------------------
// Problem constants
//   x:[8,512,64,64] f32, style:[8,512], weight:[1,512,512,3,3],
//   mod_weight:[512,512], mod_bias:[512]  ->  out:[8,512,128,128] f32
// ---------------------------------------------------------------------------
#define LIN_SCALE   0.04419417382415922f     // 1/sqrt(512)
#define CONV_SCALE  0.014731391274719739f    // 1/sqrt(512*9)
#define CS2         (1.0f / 4608.0f)         // conv_scale^2

// workspace layout (bytes)
#define OFF_S      0            //  8*512 f32
#define OFF_DEMOD  16384        //  8*512 f32
#define OFF_WSQ    32768        //  512*512 f32
#define OFF_X16R   1081344      //  8*16*66*66*32 f16 = 35,684,352 B
#define OFF_CEFF   36765696     //  8*4*9*512*512 f16 = 150,994,944 B

// ---------------------------------------------------------------------------
// s[b][i] = (style[b] . mod_weight[i]) * lin_scale + mod_bias[i]
// ---------------------------------------------------------------------------
__global__ void k_style(const float* __restrict__ style, const float* __restrict__ mw,
                        const float* __restrict__ mb, float* __restrict__ s) {
  int idx = blockIdx.x * 256 + threadIdx.x;      // 4096 exact
  int b = idx >> 9, i = idx & 511;
  const float* st = style + b * 512;
  const float* wr = mw + (size_t)i * 512;
  float acc = 0.f;
  for (int j = 0; j < 512; ++j) acc += st[j] * wr[j];
  s[idx] = acc * LIN_SCALE + mb[i];
}

// wsq[o][i] = sum_kk weight[o,i,kk]^2
__global__ void k_wsq(const float* __restrict__ w, float* __restrict__ wsq) {
  int idx = blockIdx.x * 256 + threadIdx.x;      // 262144 exact
  const float* p = w + (size_t)idx * 9;
  float a = 0.f;
  #pragma unroll
  for (int t = 0; t < 9; ++t) a += p[t] * p[t];
  wsq[idx] = a;
}

// demod[b][o] = rsqrt(cs2 * sum_i s[b,i]^2 * wsq[o,i] + eps)
__global__ void k_demod(const float* __restrict__ s, const float* __restrict__ wsq,
                        float* __restrict__ demod) {
  int idx = blockIdx.x * 256 + threadIdx.x;      // 4096 exact
  int b = idx >> 9, o = idx & 511;
  const float* sb = s + b * 512;
  const float* wr = wsq + (size_t)o * 512;
  float a = 0.f;
  for (int i = 0; i < 512; ++i) { float si = sb[i]; a += si * si * wr[i]; }
  demod[idx] = rsqrtf(CS2 * a + 1e-8f);
}

// ---------------------------------------------------------------------------
// x (f32, [b][c][64][64]) -> f16 padded+repacked [b][chunk16][y66][x66][k32]
// (k innermost -> per-lane WMMA B-fragment slice is one contiguous 32B load)
// ---------------------------------------------------------------------------
__global__ void k_xpack(const float* __restrict__ x, _Float16* __restrict__ xr) {
  int t  = blockIdx.x * 256 + threadIdx.x;       // 17,842,176 exact
  int k  = t & 31;
  int t2 = t >> 5;
  int xc = t2 % 66; t2 /= 66;
  int yc = t2 % 66; t2 /= 66;
  int chunk = t2 & 15;
  int b = t2 >> 4;
  int iy = yc - 1, ix = xc - 1;
  float v = 0.f;
  if ((unsigned)iy < 64u && (unsigned)ix < 64u)
    v = x[(((size_t)b * 512 + chunk * 32 + k) * 64 + iy) * 64 + ix];
  xr[t] = (_Float16)v;
}

// ---------------------------------------------------------------------------
// ceff, pre-swizzled into WMMA 16-bit A-fragment order.
// Element layout: [(((b*4+cls)*9+tap)*32+mtile)*16+kchunk][lane32][j16]
// with A-layout k(j): lanes0-15 hold K=0..15 ; lanes16-31 hold K=16..31
// ---------------------------------------------------------------------------
__global__ void k_ceff(const float* __restrict__ w, const float* __restrict__ s,
                       const float* __restrict__ demod, _Float16* __restrict__ cf) {
  int t = blockIdx.x * 256 + threadIdx.x;        // 75,497,472 exact
  int j      = t & 15;
  int lane   = (t >> 4) & 31;
  int kchunk = (t >> 9) & 15;
  int mtile  = (t >> 13) & 31;
  int hi   = t >> 18;
  int tap  = hi % 9;
  int hi9  = hi / 9;
  int cls  = hi9 & 3;
  int b    = hi9 >> 2;

  int m = lane & 15, h = lane >> 4;
  int k = j + 16 * h;   // contiguous k order (matches B fragment)
  int o = mtile * 16 + m;
  int i = kchunk * 32 + k;
  int ay = tap / 3, ax = tap % 3;
  int ry = cls >> 1, rx = cls & 1;

  const float kv[4] = {0.25f, 0.75f, 0.75f, 0.25f};  // separable blur taps
  float gy[3], gx[3];
  #pragma unroll
  for (int ky = 0; ky < 3; ++ky) {
    int dy = ky + 2 * ay - 1 - ry;
    gy[ky] = (dy >= 0 && dy < 4) ? kv[dy] : 0.f;
    int dx = ky + 2 * ax - 1 - rx;
    gx[ky] = (dx >= 0 && dx < 4) ? kv[dx] : 0.f;
  }
  const float* wp = w + ((size_t)o * 512 + i) * 9;
  float v = 0.f;
  #pragma unroll
  for (int ky = 0; ky < 3; ++ky)
    #pragma unroll
    for (int kx = 0; kx < 3; ++kx)
      v += gy[ky] * gx[kx] * wp[ky * 3 + kx];
  float scale = CONV_SCALE * s[b * 512 + i] * demod[b * 512 + o];
  cf[t] = (_Float16)(v * scale);
}

// ---------------------------------------------------------------------------
// Main fused kernel: 4 parity-class 3x3 convs as WMMA GEMMs sharing x patches.
// Block = 256 thr = 8 waves (4 M-groups x 2 N-groups).
// Wave: 2 M-tiles x 2 N-tiles x 4 classes = 16 f32 accumulators.
// Grid: (64 N-blocks of 4row x 16col, 4 M-blocks of 128 out-ch, 8 batch)
// ---------------------------------------------------------------------------
__global__ __launch_bounds__(256) void k_main(const _Float16* __restrict__ x16r,
                                              const _Float16* __restrict__ ceffw,
                                              float* __restrict__ out) {
  const int lane   = threadIdx.x & 31;
  const int wid    = threadIdx.x >> 5;
  const int wave_m = wid >> 1;                   // 0..3
  const int wave_n = wid & 1;                    // 0..1
  const int b      = blockIdx.z;
  const int mblk   = blockIdx.y;                 // 0..3
  const int nb     = blockIdx.x;                 // 0..63
  const int rowbase = (nb >> 2) * 4;             // Y0 grid rows
  const int colbase = (nb & 3) * 16;             // Y0 grid cols
  const int col  = lane & 15;
  const int half = lane >> 4;

  v8f acc[2][2][4];
  #pragma unroll
  for (int tm = 0; tm < 2; ++tm)
    #pragma unroll
    for (int tn = 0; tn < 2; ++tn)
      #pragma unroll
      for (int c = 0; c < 4; ++c)
        acc[tm][tn][c] = (v8f){0.f, 0.f, 0.f, 0.f, 0.f, 0.f, 0.f, 0.f};

  const int yr0    = rowbase + wave_n * 2;
  const int mtile0 = mblk * 8 + wave_m * 2;
  const _Float16* xb = x16r + (size_t)b * (16 * 4356 * 32);

  for (int kc = 0; kc < 16; ++kc) {
    const _Float16* xc = xb + (size_t)kc * (4356 * 32);
    if (kc + 1 < 16)   // pull next k-chunk's x tile toward L2/L0
      __builtin_prefetch(xc + (size_t)(4356 * 32) +
                         (((size_t)(yr0 * 66 + colbase + col)) << 5), 0, 1);
    #pragma unroll
    for (int tap = 0; tap < 9; ++tap) {
      const int ay = tap / 3, ax = tap % 3;
      v16h Bf[2];
      #pragma unroll
      for (int tn = 0; tn < 2; ++tn) {
        const int prow = yr0 + tn + ay;          // padded x row  (Y0 + a)
        const int pcol = colbase + col + ax;     // padded x col
        Bf[tn] = *(const v16h*)(xc + (((size_t)(prow * 66 + pcol)) << 5) + (half << 4));
      }
      #pragma unroll
      for (int cls = 0; cls < 4; ++cls) {
        #pragma unroll
        for (int tm = 0; tm < 2; ++tm) {
          const size_t fi =
              ((((size_t)(b * 4 + cls) * 9 + tap) * 32) + (mtile0 + tm)) * 16 + kc;
          const v16h Af = *(const v16h*)(ceffw + (fi << 9) + (lane << 4));
          #pragma unroll
          for (int tn = 0; tn < 2; ++tn)
            acc[tm][tn][cls] = __builtin_amdgcn_wmma_f32_16x16x32_f16(
                false, Af, false, Bf[tn], (short)0, acc[tm][tn][cls], false, false);
        }
      }
    }
  }

  // scatter: class (ry,rx) -> final pixel (2*Y0+ry, 2*X0+rx); pack rx pair as float2
  const int xp = 2 * (colbase + col);
  #pragma unroll
  for (int tm = 0; tm < 2; ++tm) {
    const int obase = mblk * 128 + wave_m * 32 + tm * 16 + half * 8;
    #pragma unroll
    for (int tn = 0; tn < 2; ++tn) {
      const int Y0 = yr0 + tn;
      #pragma unroll
      for (int ry = 0; ry < 2; ++ry) {
        const int y = 2 * Y0 + ry;
        const v8f a0 = acc[tm][tn][ry * 2 + 0];
        const v8f a1 = acc[tm][tn][ry * 2 + 1];
        #pragma unroll
        for (int r = 0; r < 8; ++r) {
          const int o = obase + r;
          float2 v = make_float2(a0[r], a1[r]);
          *(float2*)(out + (((size_t)(b * 512 + o) * 128 + y) * 128 + xp)) = v;
        }
      }
    }
  }
}

// ---------------------------------------------------------------------------
extern "C" void kernel_launch(void* const* d_in, const int* in_sizes, int n_in,
                              void* d_out, int out_size, void* d_ws, size_t ws_size,
                              hipStream_t stream) {
  const float* x          = (const float*)d_in[0];
  const float* style      = (const float*)d_in[1];
  const float* weight     = (const float*)d_in[2];
  const float* mod_weight = (const float*)d_in[3];
  const float* mod_bias   = (const float*)d_in[4];
  float* out = (float*)d_out;

  char* ws = (char*)d_ws;
  float*    s     = (float*)(ws + OFF_S);
  float*    demod = (float*)(ws + OFF_DEMOD);
  float*    wsq   = (float*)(ws + OFF_WSQ);
  _Float16* x16r  = (_Float16*)(ws + OFF_X16R);
  _Float16* ceffw = (_Float16*)(ws + OFF_CEFF);

  k_style<<<16,     256, 0, stream>>>(style, mod_weight, mod_bias, s);
  k_wsq  <<<1024,   256, 0, stream>>>(weight, wsq);
  k_demod<<<16,     256, 0, stream>>>(s, wsq, demod);
  k_xpack<<<69696,  256, 0, stream>>>(x, x16r);
  k_ceff <<<294912, 256, 0, stream>>>(weight, s, demod, ceffw);
  k_main <<<dim3(64, 4, 8), 256, 0, stream>>>(x16r, ceffw, out);
}
